// RASameFlop_84825604096725
// MI455X (gfx1250) — matrix operand order repaired and checked by reading, not compiled, not run on hardware
//
#include <hip/hip_runtime.h>

// ---------------- problem constants ----------------
#define B_    2
#define H_    12
#define T_    2048
#define D_    64
#define R_    16
#define DSTD_ 48
#define QTILE 128   // query rows per block (8 waves x 16 rows)
#define KTILE 32    // keys per inner step

#define AS1 __attribute__((address_space(1)))
#define AS3 __attribute__((address_space(3)))

typedef __attribute__((ext_vector_type(16))) __bf16 v16bf;
typedef __attribute__((ext_vector_type(8)))  float  v8f;
typedef __attribute__((ext_vector_type(4)))  int    v4i;

union Frag16 { v16bf v; unsigned u[8]; };

__device__ __forceinline__ unsigned short f2bf(float f) {
  unsigned u = __float_as_uint(f);
  unsigned r = u + 0x7FFFu + ((u >> 16) & 1u);   // round-to-nearest-even
  return (unsigned short)(r >> 16);
}

// A-operand (16x32, 16-bit) per-lane dword(pair) index inside a 32-elem K chunk:
// lanes 0-15 : vgpr 0-3 -> pairs 0..3  (K 0..7),  vgpr 4-7 -> pairs 8..11 (K 16..23)
// lanes 16-31: vgpr 0-3 -> pairs 4..7  (K 8..15), vgpr 4-7 -> pairs 12..15 (K 24..31)
__device__ __forceinline__ int pidxA(int hi, int v) {
  return (v & 3) + ((v >> 2) << 3) + (hi << 2);
}
// B-operand (32x16, 16-bit) per-lane dword(pair) index:
// same K order as A: lanes 0-15 hold K 0..7,16..23; lanes 16-31 hold K 8..15,24..31
__device__ __forceinline__ int pidxB(int hi, int v) {
  return (v & 3) + ((v >> 2) << 3) + (hi << 2);
}

// ---- async global->LDS 16B copy (ASYNCcnt-tracked) with portable fallback ----
__device__ __forceinline__ void async_copy16(const void* g, void* l) {
#if defined(__has_builtin) && __has_builtin(__builtin_amdgcn_global_load_async_to_lds_b128)
  __builtin_amdgcn_global_load_async_to_lds_b128(
      (AS1 v4i*)(uintptr_t)g, (AS3 v4i*)(unsigned)(uintptr_t)l, 0, 0);
#else
  // generic LDS address low 32 bits == LDS byte offset (ISA 10.2 aperture rules)
  unsigned lofs = (unsigned)(uintptr_t)l;
  asm volatile("global_load_async_to_lds_b128 %0, %1, off"
               :: "v"(lofs), "v"(g) : "memory");
#endif
}

__device__ __forceinline__ void wait_async0() {
#if defined(__has_builtin) && __has_builtin(__builtin_amdgcn_s_wait_asynccnt)
  __builtin_amdgcn_s_wait_asynccnt(0);
#else
  asm volatile("s_wait_asynccnt 0" ::: "memory");
#endif
}

// ---------------------------------------------------------------------------
// Kernel 1: build bf16 Q_aug, K_aug (row-major), transposed bf16 V_t[B,H,D,T],
// and pre-scaled bias w_disc[h]*d_bias[b,h,t].
// grid = B*H*T blocks, 64 threads.
// ---------------------------------------------------------------------------
__global__ __launch_bounds__(64) void prep_kernel(
    const float* __restrict__ Q, const float* __restrict__ K,
    const float* __restrict__ V, const float* __restrict__ d_bias,
    const float* __restrict__ W, const float* __restrict__ w_std,
    const float* __restrict__ w_rec, const float* __restrict__ w_disc,
    unsigned short* __restrict__ Qa, unsigned short* __restrict__ Ka,
    unsigned short* __restrict__ Vt, float* __restrict__ biasArr)
{
  const int n  = blockIdx.x;        // flat (b,h,t)
  const int t  = n % T_;
  const int bh = n / T_;
  const int h  = bh % H_;
  const int d  = threadIdx.x;       // 0..63

  __shared__ float qs[D_], ks[D_], ql[R_], kl[R_];

  const float qv = Q[(size_t)n * D_ + d];
  const float kv = K[(size_t)n * D_ + d];
  qs[d] = qv;
  ks[d] = kv;
  __syncthreads();

  if (d < R_) {
    float aq = 0.f, ak = 0.f;
#pragma unroll 8
    for (int i = 0; i < D_; ++i) {
      const float w = W[i * R_ + d];
      aq += qs[i] * w;
      ak += ks[i] * w;
    }
    ql[d] = aq;
    kl[d] = ak;
  }
  __syncthreads();

  const float sstd = sqrtf(w_std[h]);
  const float srec = sqrtf(w_rec[h]);
  const float qa = (d < DSTD_) ? sstd * qv : srec * kl[d - DSTD_];
  const float ka = (d < DSTD_) ? sstd * kv : srec * ql[d - DSTD_];

  Qa[(size_t)n * D_ + d] = f2bf(qa);
  Ka[(size_t)n * D_ + d] = f2bf(ka);
  Vt[((size_t)bh * D_ + d) * T_ + t] = f2bf(V[(size_t)n * D_ + d]);
  if (d == 0)
    biasArr[(size_t)bh * T_ + t] = w_disc[h] * d_bias[(size_t)bh * T_ + t];
}

// ---- softmax/rescale step for one 16x32 score tile pair -------------------
template <bool MASK>
__device__ __forceinline__ void softmax_step(
    const v8f& s0v, const v8f& s1v, float b0, float b1,
    int kb, int qrow0, int hi, int col,
    float (&m)[8], float (&l)[8], v8f (&acc)[4],
    unsigned short (*pw)[KTILE])
{
#pragma unroll
  for (int j = 0; j < 8; ++j) {
    const int qrow = qrow0 + j + (hi ? 8 : 0);
    float s0 = s0v[j] * 0.125f + b0;      // 1/sqrt(D) = 0.125
    float s1 = s1v[j] * 0.125f + b1;
    if (MASK) {
      if (kb + col > qrow)      s0 = -1e30f;   // causal mask
      if (kb + 16 + col > qrow) s1 = -1e30f;
    }
    float mx = fmaxf(s0, s1);
#pragma unroll
    for (int o = 8; o >= 1; o >>= 1) mx = fmaxf(mx, __shfl_xor(mx, o, 32));
    const float mnew = fmaxf(m[j], mx);
    const float f  = __expf(m[j] - mnew);
    const float p0 = __expf(s0 - mnew);
    const float p1 = __expf(s1 - mnew);
    float rs = p0 + p1;
#pragma unroll
    for (int o = 8; o >= 1; o >>= 1) rs += __shfl_xor(rs, o, 32);
    l[j] = l[j] * f + rs;
    m[j] = mnew;
#pragma unroll
    for (int dt = 0; dt < 4; ++dt) acc[dt][j] *= f;
    const int pr = j + (hi ? 8 : 0);
    pw[pr][col]      = f2bf(p0);
    pw[pr][col + 16] = f2bf(p1);
  }
}

// ---------------------------------------------------------------------------
// Kernel 2: causal flash attention. K/V tiles staged in LDS via async-to-LDS
// DMA, double buffered; WMMA bf16 matmuls with fp32 accumulation.
// grid = (T/QTILE, B*H), 256 threads = 8 waves; wave w owns 16 query rows.
// ---------------------------------------------------------------------------
__global__ __launch_bounds__(256) void fa_kernel(
    const unsigned short* __restrict__ Qa, const unsigned short* __restrict__ Ka,
    const unsigned short* __restrict__ Vt, const float* __restrict__ biasArr,
    float* __restrict__ Out)
{
  const int bh    = blockIdx.y;
  const int qt    = blockIdx.x;
  const int w     = threadIdx.x >> 5;
  const int lane  = threadIdx.x & 31;
  const int hi    = lane >> 4;
  const int col   = lane & 15;
  const int qrow0 = qt * QTILE + w * 16;

  __shared__ unsigned short ldsK[2][KTILE][D_];    // 2 x 4KB (32 keys x 64 dims)
  __shared__ unsigned short ldsV[2][D_][KTILE];    // 2 x 4KB (64 dims x 32 keys)
  __shared__ unsigned short lds_p[8][16][KTILE];   // 8KB per-wave P tiles

  const unsigned short* Kbh = Ka + (size_t)bh * T_ * D_;
  const unsigned short* Vbh = Vt + (size_t)bh * D_ * T_;
  const float*          bias = biasArr + (size_t)bh * T_;

  // cooperative async fill of one K/V double-buffer slot
  auto load_tiles = [&](int b, int kb) {
    {
      const int r = threadIdx.x >> 3, seg = threadIdx.x & 7;   // 32 rows x 8 segs
      async_copy16(Kbh + (size_t)(kb + r) * D_ + seg * 8, &ldsK[b][r][seg * 8]);
    }
    {
      const int r = threadIdx.x >> 2, seg = threadIdx.x & 3;   // 64 rows x 4 segs
      async_copy16(Vbh + (size_t)r * T_ + kb + seg * 8, &ldsV[b][r][seg * 8]);
    }
  };

  // ---- load Q_aug A-fragments (two 32-wide K chunks), kept for whole kernel
  Frag16 aq[2];
  {
    const unsigned* rowp =
        (const unsigned*)(Qa + ((size_t)bh * T_ + qrow0 + col) * D_);
#pragma unroll
    for (int c = 0; c < 2; ++c)
#pragma unroll
      for (int v = 0; v < 8; ++v)
        aq[c].u[v] = rowp[c * 16 + pidxA(hi, v)];
  }

  float m[8], l[8];
  v8f acc[4];
#pragma unroll
  for (int j = 0; j < 8; ++j) { m[j] = -1e30f; l[j] = 0.f; }
#pragma unroll
  for (int dt = 0; dt < 4; ++dt)
    acc[dt] = (v8f){0.f, 0.f, 0.f, 0.f, 0.f, 0.f, 0.f, 0.f};

  const int nkt = qt * 4 + 4;             // causal: keys < (qt+1)*128

  int cur = 0;
  load_tiles(0, 0);
  wait_async0();
  __syncthreads();

  for (int kt = 0; kt < nkt; ++kt) {
    const int kb = kt * KTILE;

    // kick off DMA for next tile while we compute on this one
    if (kt + 1 < nkt) load_tiles(cur ^ 1, kb + KTILE);

    // ---- S = Q_aug x K_aug^T  (two 16-key sub-tiles, K-dim 64 = 2 chunks)
    v8f s[2];
#pragma unroll
    for (int st = 0; st < 2; ++st) {
      Frag16 bk[2];
      const unsigned* krow = (const unsigned*)&ldsK[cur][st * 16 + col][0];
#pragma unroll
      for (int c = 0; c < 2; ++c)
#pragma unroll
        for (int v = 0; v < 8; ++v)
          bk[c].u[v] = krow[c * 16 + pidxB(hi, v)];

      v8f z = (v8f){0.f, 0.f, 0.f, 0.f, 0.f, 0.f, 0.f, 0.f};
      z = __builtin_amdgcn_wmma_f32_16x16x32_bf16(false, aq[0].v, false, bk[0].v,
                                                  (short)0, z, false, false);
      z = __builtin_amdgcn_wmma_f32_16x16x32_bf16(false, aq[1].v, false, bk[1].v,
                                                  (short)0, z, false, false);
      s[st] = z;
    }

    const float b0 = bias[kb + col];
    const float b1 = bias[kb + 16 + col];

    // ---- online softmax; diagonal-block tiles take the masked path
    if (kb + KTILE > qrow0)
      softmax_step<true >(s[0], s[1], b0, b1, kb, qrow0, hi, col, m, l, acc, lds_p[w]);
    else
      softmax_step<false>(s[0], s[1], b0, b1, kb, qrow0, hi, col, m, l, acc, lds_p[w]);
    __syncthreads();   // uniform trip count; fences P writes before A-layout reads

    // ---- re-load P in A-operand layout (row = lane&15)
    Frag16 pa;
    {
      const unsigned* prow = (const unsigned*)&lds_p[w][col][0];
#pragma unroll
      for (int v = 0; v < 8; ++v) pa.u[v] = prow[pidxA(hi, v)];
    }

    // ---- O += P x V   (LDS V tile is [dim][key] so key-pairs are dwords)
#pragma unroll
    for (int dt = 0; dt < 4; ++dt) {
      Frag16 bv;
      const unsigned* vrow = (const unsigned*)&ldsV[cur][dt * 16 + col][0];
#pragma unroll
      for (int v = 0; v < 8; ++v) bv.u[v] = vrow[pidxB(hi, v)];
      acc[dt] = __builtin_amdgcn_wmma_f32_16x16x32_bf16(false, pa.v, false, bv.v,
                                                        (short)0, acc[dt], false, false);
    }

    wait_async0();      // next K/V slot landed
    __syncthreads();    // all waves done reading old slot before it is refilled
    cur ^= 1;
  }

  // ---- normalize and store (C-layout: VGPR j -> row j / j+8, col = lane&15)
#pragma unroll
  for (int j = 0; j < 8; ++j) {
    const float invl = 1.0f / l[j];
    const int   row  = qrow0 + j + (hi ? 8 : 0);
    float* orow = Out + ((size_t)bh * T_ + row) * D_;
#pragma unroll
    for (int dt = 0; dt < 4; ++dt)
      orow[dt * 16 + col] = acc[dt][j] * invl;
  }
}

// ---------------------------------------------------------------------------
extern "C" void kernel_launch(void* const* d_in, const int* in_sizes, int n_in,
                              void* d_out, int out_size, void* d_ws, size_t ws_size,
                              hipStream_t stream) {
  const float* Q      = (const float*)d_in[0];
  const float* K      = (const float*)d_in[1];
  const float* V      = (const float*)d_in[2];
  const float* d_bias = (const float*)d_in[3];
  const float* W      = (const float*)d_in[4];
  const float* w_std  = (const float*)d_in[5];
  const float* w_rec  = (const float*)d_in[6];
  const float* w_disc = (const float*)d_in[7];
  float* Out = (float*)d_out;

  const size_t N = (size_t)B_ * H_ * T_ * D_;
  unsigned short* Qa = (unsigned short*)d_ws;
  unsigned short* Ka = Qa + N;
  unsigned short* Vt = Ka + N;
  float* biasArr = (float*)(Vt + N);

  prep_kernel<<<B_ * H_ * T_, 64, 0, stream>>>(Q, K, V, d_bias, W, w_std, w_rec,
                                               w_disc, Qa, Ka, Vt, biasArr);
  dim3 grid(T_ / QTILE, B_ * H_);
  fa_kernel<<<grid, 256, 0, stream>>>(Qa, Ka, Vt, biasArr, Out);
}
